// ResidueTransformerMLP_90632399880755
// MI455X (gfx1250) — matrix operand order repaired and checked by reading, not compiled, not run on hardware
//
#include <hip/hip_runtime.h>
#include <hip/hip_bf16.h>
#include <stdint.h>

namespace {
constexpr int Bc = 16, Lc = 1022, Ec = 1280, Dc = 512, Hc = 8, FFc = 2048, NLc = 4, DHc = 64;
constexpr long Mc  = (long)Bc * Lc;   // 16352 rows, multiple of 16
constexpr int  LPc = 1024;            // keys padded to multiple of 32
constexpr float EPSc = 1e-5f;
}

typedef __attribute__((ext_vector_type(16))) _Float16 v16h;
typedef __attribute__((ext_vector_type(8)))  float    v8f;
typedef int vint4 __attribute__((vector_size(16)));   // matches builtin's int4 pointee

// ---------------- CDNA5 async global->LDS copy (ASYNCcnt-tracked) ----------------
#if defined(__has_builtin)
#if __has_builtin(__builtin_amdgcn_global_load_async_to_lds_b128)
#define HAVE_ASYNC_BUILTIN 1
#endif
#if __has_builtin(__builtin_amdgcn_s_wait_asynccnt)
#define HAVE_WAITASYNC_BUILTIN 1
#endif
#endif

__device__ __forceinline__ void async_copy_b128(const _Float16* g, _Float16* l) {
#if defined(HAVE_ASYNC_BUILTIN)
  __builtin_amdgcn_global_load_async_to_lds_b128(
      (__attribute__((address_space(1))) vint4*)(uintptr_t)g,
      (__attribute__((address_space(3))) vint4*)(uint32_t)(uintptr_t)l, 0, 0);
#else
  const uint32_t loff = (uint32_t)(uintptr_t)l;  // low 32 bits of generic LDS ptr = LDS offset
  asm volatile("global_load_async_to_lds_b128 %0, %1, off" :: "v"(loff), "v"(g) : "memory");
#endif
}

#if defined(HAVE_WAITASYNC_BUILTIN)
#define WAIT_ASYNC(n) __builtin_amdgcn_s_wait_asynccnt(n)
#else
#define WAIT_ASYNC(n) asm volatile("s_wait_asynccnt " #n ::: "memory")
#endif

// ---------------- WMMA fragment loaders (wave32, V_WMMA_F32_16X16X32_F16) ----------------
// A (16x32 f16): lanes 0-15 -> M=lane, kbase=0 ; lanes 16-31 -> M=lane-16, kbase=16
// VGPR 0..7 hold K = kbase+{0..15} (2 halves per VGPR)
__device__ __forceinline__ v16h frag_a(const _Float16* __restrict__ p, int ld) {
  const int lane = threadIdx.x & 31;
  const _Float16* q = p + (long)(lane & 15) * ld + ((lane >> 4) << 4);
  v16h f;
#pragma unroll
  for (int r = 0; r < 8; ++r) {
    const int ko = 2 * r;
    f[2 * r]     = q[ko];
    f[2 * r + 1] = q[ko + 1];
  }
  return f;
}
// B (32x16 f16) from row-major W[N,K]: lanes 0-15 -> N=lane, K=0..15 ;
// lanes 16-31 -> N=lane-16, K=16..31 (16 contiguous halves per lane)
__device__ __forceinline__ v16h frag_b(const _Float16* __restrict__ p, int ld) {
  const int lane = threadIdx.x & 31;
  const _Float16* q = p + (long)(lane & 15) * ld + ((lane >> 4) << 4);
  v16h f;
#pragma unroll
  for (int j = 0; j < 16; ++j) f[j] = q[j];
  return f;
}

// ---------------- generic elementwise f32 -> f16 ----------------
__global__ void cvt_f16_kernel(const float* __restrict__ in, _Float16* __restrict__ out, long n) {
  long i = (long)blockIdx.x * blockDim.x + threadIdx.x;
  const long stride = (long)gridDim.x * blockDim.x;
  for (; i < n; i += stride) out[i] = (_Float16)in[i];
}

// ---------------- WMMA GEMM: C[M,N] = A[M,K](f16) * W[N,K]^T(f16) + bias (+resid)(relu) ----
// Block = 4 waves, 64 (M) x 64 (N) tile. Each wave: 16x64 (4 accumulators, A frag reused 4x).
// B tile (64x32 f16 = 4KB) staged to LDS via async global->LDS, double-buffered.
__global__ __launch_bounds__(128) void gemm_wmma_kernel(
    const _Float16* __restrict__ A, int lda,
    const _Float16* __restrict__ W, int ldw,
    const float* __restrict__ bias, const float* __restrict__ residual,
    float* __restrict__ Cf, _Float16* __restrict__ Ch,
    int M, int N, int K, int flags /*bit0: relu*/) {
  __shared__ _Float16 bs[2][64 * 32];            // [buf][n(64)][k(32)]
  const int tid  = threadIdx.x;
  const int wave = tid >> 5;
  const int lane = tid & 31;
  const int m0 = blockIdx.y * 64 + wave * 16;
  const int n0 = blockIdx.x * 64;
  const int m0c = (m0 < M) ? m0 : (M - 16);      // clamp tail (no early return: barriers)
  const _Float16* Ap = A + (long)m0c * lda;

  // staging map: thread t copies 32B of row (t>>1), k-halves [(t&1)*16 .. +16)
  const int brow = tid >> 1;
  const int bk   = (tid & 1) << 4;
  const _Float16* Wg = W + (long)(n0 + brow) * ldw + bk;

  auto issue = [&](int kk, int buf) {
    _Float16* l = &bs[buf][brow * 32 + bk];
    async_copy_b128(Wg + kk, l);
    async_copy_b128(Wg + kk + 8, l + 8);
  };

  v8f acc[4] = {{}, {}, {}, {}};
  const int T = K / 32;
  issue(0, 0);
  for (int t = 0; t < T; ++t) {
    const int buf = t & 1;
    if (t + 1 < T) {
      issue((t + 1) * 32, buf ^ 1);
      WAIT_ASYNC(2);                             // tile t done; tile t+1 in flight
    } else {
      WAIT_ASYNC(0);
    }
    __syncthreads();                             // all waves' copies of tile t visible
    const v16h a = frag_a(Ap + t * 32, lda);
#pragma unroll
    for (int g = 0; g < 4; ++g) {
      const _Float16* bp = &bs[buf][(g * 16 + (lane & 15)) * 32 + ((lane >> 4) << 4)];
      v16h b;
#pragma unroll
      for (int j = 0; j < 16; ++j) b[j] = bp[j];
      acc[g] = __builtin_amdgcn_wmma_f32_16x16x32_f16(false, a, false, b, (short)0, acc[g], false, false);
    }
    __syncthreads();                             // reads done before buf is overwritten
  }

  if (m0 < M) {
    const int cn = lane & 15;
    const int mb = (lane >> 4) << 3;
#pragma unroll
    for (int g = 0; g < 4; ++g) {
      const int n = n0 + g * 16 + cn;
      const float bv = bias ? bias[n] : 0.0f;
#pragma unroll
      for (int v = 0; v < 8; ++v) {
        const long idx = (long)(m0 + mb + v) * N + n;
        float val = acc[g][v] + bv;
        if (residual) val += residual[idx];
        if (flags & 1) val = fmaxf(val, 0.0f);
        if (Cf) Cf[idx] = val;
        if (Ch) Ch[idx] = (_Float16)val;
      }
    }
  }
}

// ---------------- LayerNorm (one row per 128-thread block) ----------------
__global__ __launch_bounds__(128) void ln_kernel(
    const float* __restrict__ X, const float* __restrict__ s, const float* __restrict__ bsh,
    float* __restrict__ Of, _Float16* __restrict__ Oh, int Dn, int relu) {
  __shared__ float red[128];
  __shared__ float stats[2];
  const long row = blockIdx.x;
  const float* x = X + row * (long)Dn;
  float ls = 0.f;
  for (int d = threadIdx.x; d < Dn; d += 128) ls += x[d];
  red[threadIdx.x] = ls; __syncthreads();
  for (int o = 64; o > 0; o >>= 1) { if ((int)threadIdx.x < o) red[threadIdx.x] += red[threadIdx.x + o]; __syncthreads(); }
  if (threadIdx.x == 0) stats[0] = red[0] / Dn;
  __syncthreads();
  const float mean = stats[0];
  float lq = 0.f;
  for (int d = threadIdx.x; d < Dn; d += 128) { float v = x[d] - mean; lq += v * v; }
  red[threadIdx.x] = lq; __syncthreads();
  for (int o = 64; o > 0; o >>= 1) { if ((int)threadIdx.x < o) red[threadIdx.x] += red[threadIdx.x + o]; __syncthreads(); }
  if (threadIdx.x == 0) stats[1] = rsqrtf(red[0] / Dn + EPSc);
  __syncthreads();
  const float rstd = stats[1];
  for (int d = threadIdx.x; d < Dn; d += 128) {
    float v = (x[d] - mean) * rstd * s[d] + bsh[d];
    if (relu) v = fmaxf(v, 0.f);
    if (Of) Of[row * (long)Dn + d] = v;
    if (Oh) Oh[row * (long)Dn + d] = (_Float16)v;
  }
}

// ---------------- LayerNorm + RoPE: writes n (f16) and rope(n) (f16) ----------------
__global__ __launch_bounds__(128) void ln_rope_kernel(
    const float* __restrict__ X, const float* __restrict__ s, const float* __restrict__ bsh,
    _Float16* __restrict__ Nh, _Float16* __restrict__ Rh) {
  __shared__ float red[128];
  __shared__ float stats[2];
  __shared__ float nr[Dc];
  const long row = blockIdx.x;
  const int l = (int)(row % Lc);
  const float* x = X + row * (long)Dc;
  float ls = 0.f;
  for (int d = threadIdx.x; d < Dc; d += 128) ls += x[d];
  red[threadIdx.x] = ls; __syncthreads();
  for (int o = 64; o > 0; o >>= 1) { if ((int)threadIdx.x < o) red[threadIdx.x] += red[threadIdx.x + o]; __syncthreads(); }
  if (threadIdx.x == 0) stats[0] = red[0] / Dc;
  __syncthreads();
  const float mean = stats[0];
  float lq = 0.f;
  for (int d = threadIdx.x; d < Dc; d += 128) { float v = x[d] - mean; lq += v * v; }
  red[threadIdx.x] = lq; __syncthreads();
  for (int o = 64; o > 0; o >>= 1) { if ((int)threadIdx.x < o) red[threadIdx.x] += red[threadIdx.x + o]; __syncthreads(); }
  if (threadIdx.x == 0) stats[1] = rsqrtf(red[0] / Dc + EPSc);
  __syncthreads();
  const float rstd = stats[1];
  for (int d = threadIdx.x; d < Dc; d += 128) {
    float v = (x[d] - mean) * rstd * s[d] + bsh[d];
    nr[d] = v;
    Nh[row * (long)Dc + d] = (_Float16)v;
  }
  __syncthreads();
  for (int d = threadIdx.x; d < Dc; d += 128) {
    const int j = d & (DHc - 1);         // index within head
    const int i = j & 31;                // frequency index
    const float inv = __expf(-(2.0f * (float)i / (float)DHc) * 9.210340372f); // 10000^(-2i/DH)
    const float ang = (float)l * inv;
    const float c = __cosf(ang), sn = __sinf(ang);
    const float rot = (j < 32) ? -nr[d + 32] : nr[d - 32];
    Rh[row * (long)Dc + d] = (_Float16)(nr[d] * c + rot * sn);
  }
}

// ---------------- Attention: one wave per (b, h, 16-query tile) ----------------
__global__ __launch_bounds__(32) void attn_kernel(
    const _Float16* __restrict__ Q, const _Float16* __restrict__ Kc,
    const _Float16* __restrict__ V, const float* __restrict__ mask,
    _Float16* __restrict__ O) {
  __shared__ float sc[16 * LPc];         // 64 KB score/prob buffer
  const int qt = blockIdx.x, h = blockIdx.y, b = blockIdx.z;
  const int lane = threadIdx.x;
  const int hoff = h * DHc;
  const long base = (long)b * Lc * Dc;
  const float scale = 0.125f;            // 1/sqrt(64)

  const _Float16* qp = Q + base + (long)(qt * 16) * Dc + hoff;
  const v16h qa0 = frag_a(qp, Dc);
  const v16h qa1 = frag_a(qp + 32, Dc);

  for (int kt = 0; kt < LPc / 16; ++kt) {
    const _Float16* kp = Kc + base + (long)(kt * 16) * Dc + hoff;
    v8f accs = {};
    accs = __builtin_amdgcn_wmma_f32_16x16x32_f16(false, qa0, false, frag_b(kp, Dc),      (short)0, accs, false, false);
    accs = __builtin_amdgcn_wmma_f32_16x16x32_f16(false, qa1, false, frag_b(kp + 32, Dc), (short)0, accs, false, false);
    const int n = lane & 15;
    const int key = kt * 16 + n;
    const float mk = (key < Lc) ? mask[b * Lc + key] : 0.0f;
    const bool valid = (key < Lc) && (mk > 0.0f);
    const int mb = (lane >> 4) << 3;
#pragma unroll
    for (int v = 0; v < 8; ++v)
      sc[(mb + v) * LPc + key] = valid ? accs[v] * scale : -1e30f;
  }

  for (int r = 0; r < 16; ++r) {
    float mx = -1e30f;
    for (int c = lane; c < LPc; c += 32) mx = fmaxf(mx, sc[r * LPc + c]);
#pragma unroll
    for (int o = 16; o > 0; o >>= 1) mx = fmaxf(mx, __shfl_xor(mx, o, 32));
    float sm = 0.f;
    for (int c = lane; c < LPc; c += 32) { float e = __expf(sc[r * LPc + c] - mx); sc[r * LPc + c] = e; sm += e; }
#pragma unroll
    for (int o = 16; o > 0; o >>= 1) sm += __shfl_xor(sm, o, 32);
    const float inv = 1.0f / sm;
    for (int c = lane; c < LPc; c += 32) sc[r * LPc + c] *= inv;
  }

  const int prow = lane & 15;
  const int kba  = (lane >> 4) << 4;     // A-layout k-base (0 or 16)
  const int kbb  = (lane >> 4) << 4;     // B-layout k-base (0 or 16)
  for (int g = 0; g < 4; ++g) {
    const int ncol = hoff + g * 16 + (lane & 15);
    v8f acc = {};
    for (int kt = 0; kt < LPc; kt += 32) {
      v16h pa;
#pragma unroll
      for (int r = 0; r < 8; ++r) {
        const int ko = 2 * r;
        pa[2 * r]     = (_Float16)sc[prow * LPc + kt + kba + ko];
        pa[2 * r + 1] = (_Float16)sc[prow * LPc + kt + kba + ko + 1];
      }
      v16h vb;
#pragma unroll
      for (int j = 0; j < 16; ++j) {
        const int kk = kt + kbb + j;
        const int kr = (kk < Lc) ? kk : (Lc - 1);   // clamp; P==0 there anyway
        vb[j] = V[base + (long)kr * Dc + ncol];
      }
      acc = __builtin_amdgcn_wmma_f32_16x16x32_f16(false, pa, false, vb, (short)0, acc, false, false);
    }
    const int n = lane & 15;
    const int mb = (lane >> 4) << 3;
#pragma unroll
    for (int v = 0; v < 8; ++v) {
      const int m = qt * 16 + mb + v;
      if (m < Lc) O[base + (long)m * Dc + hoff + g * 16 + n] = (_Float16)acc[v];
    }
  }
}

// ---------------- masked mean pool over L ----------------
__global__ __launch_bounds__(256) void pool_kernel(const float* __restrict__ X,
                                                   const float* __restrict__ mask,
                                                   float* __restrict__ G) {
  const int b = blockIdx.x;
  const int d = threadIdx.x;
  float s0 = 0.f, s1 = 0.f, ms = 0.f;
  for (int l = 0; l < Lc; ++l) {
    const float mk = mask[b * Lc + l];
    ms += mk;
    const float* xp = X + ((long)b * Lc + l) * Dc;
    s0 += xp[d] * mk;
    s1 += xp[d + 256] * mk;
  }
  G[b * Dc + d] = s0 / ms;
  G[b * Dc + d + 256] = s1 / ms;
}

// ---------------- combined = concat(x, gctx) as f16 ----------------
__global__ __launch_bounds__(256) void combined_kernel(const float* __restrict__ X,
                                                       const float* __restrict__ G,
                                                       _Float16* __restrict__ C) {
  const long r = blockIdx.x;
  const int b = (int)(r / Lc);
  for (int j = threadIdx.x; j < Dc; j += 256) {
    C[r * (2 * Dc) + j]      = (_Float16)X[r * Dc + j];
    C[r * (2 * Dc) + Dc + j] = (_Float16)G[b * Dc + j];
  }
}

// ---------------- final logits: per-row dot(h, h2_w) + b ----------------
__global__ __launch_bounds__(32) void logits_kernel(const _Float16* __restrict__ Hh,
                                                    const float* __restrict__ w,
                                                    const float* __restrict__ bb,
                                                    float* __restrict__ out) {
  const long r = blockIdx.x;
  const int lane = threadIdx.x;
  float s = 0.f;
  for (int j = lane; j < 128; j += 32) s += (float)Hh[r * 128 + j] * w[j];
#pragma unroll
  for (int o = 16; o > 0; o >>= 1) s += __shfl_xor(s, o, 32);
  if (lane == 0) out[r] = s + bb[0];
}

// =====================================================================================
extern "C" void kernel_launch(void* const* d_in, const int* in_sizes, int n_in,
                              void* d_out, int out_size, void* d_ws, size_t ws_size,
                              hipStream_t stream) {
  (void)in_sizes; (void)n_in; (void)out_size; (void)ws_size;
  const float* emb      = (const float*)d_in[0];
  const float* mask     = (const float*)d_in[1];
  const float* proj_w   = (const float*)d_in[3];
  const float* proj_b   = (const float*)d_in[4];
  const float* proj_ln_s= (const float*)d_in[5];
  const float* proj_ln_b= (const float*)d_in[6];
  const float* ln1_s    = (const float*)d_in[7];
  const float* ln1_b    = (const float*)d_in[8];
  const float* in_w     = (const float*)d_in[9];
  const float* in_b     = (const float*)d_in[10];
  const float* out_w    = (const float*)d_in[11];
  const float* out_b    = (const float*)d_in[12];
  const float* ln2_s    = (const float*)d_in[13];
  const float* ln2_b    = (const float*)d_in[14];
  const float* w1       = (const float*)d_in[15];
  const float* b1       = (const float*)d_in[16];
  const float* w2       = (const float*)d_in[17];
  const float* b2       = (const float*)d_in[18];
  const float* h1_w     = (const float*)d_in[19];
  const float* h1_b     = (const float*)d_in[20];
  const float* hln_s    = (const float*)d_in[21];
  const float* hln_b    = (const float*)d_in[22];
  const float* h2_w     = (const float*)d_in[23];
  const float* h2_b     = (const float*)d_in[24];
  float* logits = (float*)d_out;

  // ---- workspace carve-out ----
  char* ws = (char*)d_ws;
  size_t off = 0;
  auto alloc = [&](size_t bytes) -> void* {
    void* p = ws + off;
    off += (bytes + 255) & ~(size_t)255;
    return p;
  };
  const long MD = Mc * Dc, ME = Mc * Ec, MFF = Mc * (long)FFc, M2D = Mc * 2L * Dc, M128 = Mc * 128L;
  _Float16* emb16  = (_Float16*)alloc(ME * 2);
  _Float16* pw16   = (_Float16*)alloc((long)Dc * Ec * 2);
  _Float16* iw16   = (_Float16*)alloc((long)NLc * 3 * Dc * Dc * 2);
  _Float16* ow16   = (_Float16*)alloc((long)NLc * Dc * Dc * 2);
  _Float16* w116   = (_Float16*)alloc((long)NLc * FFc * Dc * 2);
  _Float16* w216   = (_Float16*)alloc((long)NLc * Dc * FFc * 2);
  _Float16* h1w16  = (_Float16*)alloc(128L * 2 * Dc * 2);
  float*    x      = (float*)   alloc(MD * 4);
  float*    y      = (float*)   alloc(MD * 4);
  _Float16* n16    = (_Float16*)alloc(MD * 2);
  _Float16* r16    = (_Float16*)alloc(MD * 2);
  _Float16* q16    = (_Float16*)alloc(MD * 2);
  _Float16* k16    = (_Float16*)alloc(MD * 2);
  _Float16* v16    = (_Float16*)alloc(MD * 2);
  _Float16* o16    = (_Float16*)alloc(MD * 2);
  _Float16* ff16   = (_Float16*)alloc(MFF * 2);
  _Float16* comb16 = (_Float16*)alloc(M2D * 2);
  float*    hf     = (float*)   alloc(M128 * 4);
  _Float16* h16    = (_Float16*)alloc(M128 * 2);
  float*    gctx   = (float*)   alloc((long)Bc * Dc * 4);

  auto cvt = [&](const float* src, _Float16* dst, long n) {
    cvt_f16_kernel<<<2048, 256, 0, stream>>>(src, dst, n);
  };
  auto gemm = [&](const _Float16* A, int lda, const _Float16* W, int ldw,
                  const float* bias, const float* resid, float* Cf, _Float16* Ch,
                  int M, int N, int K, int flags) {
    dim3 grid(N / 64, (M + 63) / 64);
    gemm_wmma_kernel<<<grid, 128, 0, stream>>>(A, lda, W, ldw, bias, resid, Cf, Ch, M, N, K, flags);
  };

  // ---- weight / input precision conversion ----
  cvt(emb,    emb16, ME);
  cvt(proj_w, pw16,  (long)Dc * Ec);
  cvt(in_w,   iw16,  (long)NLc * 3 * Dc * Dc);
  cvt(out_w,  ow16,  (long)NLc * Dc * Dc);
  cvt(w1,     w116,  (long)NLc * FFc * Dc);
  cvt(w2,     w216,  (long)NLc * Dc * FFc);
  cvt(h1_w,   h1w16, 128L * 2 * Dc);

  // ---- input projection + LN ----
  gemm(emb16, Ec, pw16, Ec, proj_b, nullptr, y, nullptr, (int)Mc, Dc, Ec, 0);
  ln_kernel<<<(int)Mc, 128, 0, stream>>>(y, proj_ln_s, proj_ln_b, x, nullptr, Dc, 0);

  // ---- transformer layers ----
  for (int i = 0; i < NLc; ++i) {
    const _Float16* wq = iw16 + (long)i * 3 * Dc * Dc;
    const _Float16* wk = wq + (long)Dc * Dc;
    const _Float16* wv = wk + (long)Dc * Dc;
    const float* bq = in_b + (long)i * 3 * Dc;
    const float* bk = bq + Dc;
    const float* bv = bk + Dc;

    ln_rope_kernel<<<(int)Mc, 128, 0, stream>>>(x, ln1_s + (long)i * Dc, ln1_b + (long)i * Dc, n16, r16);
    gemm(r16, Dc, wq, Dc, bq, nullptr, nullptr, q16, (int)Mc, Dc, Dc, 0);
    gemm(r16, Dc, wk, Dc, bk, nullptr, nullptr, k16, (int)Mc, Dc, Dc, 0);
    gemm(n16, Dc, wv, Dc, bv, nullptr, nullptr, v16, (int)Mc, Dc, Dc, 0);

    attn_kernel<<<dim3(LPc / 16, Hc, Bc), 32, 0, stream>>>(q16, k16, v16, mask, o16);

    gemm(o16, Dc, ow16 + (long)i * Dc * Dc, Dc, out_b + (long)i * Dc, x, x, nullptr, (int)Mc, Dc, Dc, 0);

    ln_kernel<<<(int)Mc, 128, 0, stream>>>(x, ln2_s + (long)i * Dc, ln2_b + (long)i * Dc, nullptr, n16, Dc, 0);
    gemm(n16, Dc, w116 + (long)i * FFc * Dc, Dc, b1 + (long)i * FFc, nullptr, nullptr, ff16, (int)Mc, FFc, Dc, 1);
    gemm(ff16, FFc, w216 + (long)i * Dc * FFc, FFc, b2 + (long)i * Dc, x, x, nullptr, (int)Mc, Dc, FFc, 0);
  }

  // ---- pooled context + head ----
  pool_kernel<<<Bc, 256, 0, stream>>>(x, mask, gctx);
  combined_kernel<<<(int)Mc, 256, 0, stream>>>(x, gctx, comb16);
  gemm(comb16, 2 * Dc, h1w16, 2 * Dc, h1_b, nullptr, hf, nullptr, (int)Mc, 128, 2 * Dc, 0);
  ln_kernel<<<(int)Mc, 128, 0, stream>>>(hf, hln_s, hln_b, nullptr, h16, 128, 1);
  logits_kernel<<<(int)Mc, 32, 0, stream>>>(h16, h2_w, h2_b, logits);
}
